// SimCLR_Loss_79972291052089
// MI455X (gfx1250) — compile-verified, matrix operand order checked by reading, not run on hardware
//
#include <hip/hip_runtime.h>
#include <hip/hip_bf16.h>
#include <math.h>

typedef __attribute__((ext_vector_type(16))) _Float16 v16h;
typedef __attribute__((ext_vector_type(8)))  _Float16 v8h;
typedef __attribute__((ext_vector_type(8)))  float    v8f;
typedef __attribute__((ext_vector_type(4)))  float    v4f;

#define NROWS   8192
#define DDIM    256
#define BHALF   4096     // B; pair(i) = i ^ BHALF (power of two)
#define NCHUNK  8        // column chunks (each 1024 columns)

static __device__ inline v16h cat16(v8h lo, v8h hi) {
  v16h r;
#pragma unroll
  for (int i = 0; i < 8; ++i) { r[i] = lo[i]; r[i + 8] = hi[i]; }
  return r;
}

// ---------------------------------------------------------------------------
// Kernel 1: L2-normalize rows of z = concat(z_i, z_j), emit f16 Zn [N, D].
// One wave per row; lane handles 8 contiguous elements (two B128 loads,
// one B128 f16 store).
// ---------------------------------------------------------------------------
__global__ __launch_bounds__(256) void simclr_normalize(
    const float* __restrict__ z_i, const float* __restrict__ z_j,
    _Float16* __restrict__ zn) {
  const int lane = threadIdx.x & 31;
  const int wave = threadIdx.x >> 5;
  const int row  = blockIdx.x * 8 + wave;            // 0..8191

  const float* src = (row < BHALF) ? (z_i + (size_t)row * DDIM)
                                   : (z_j + (size_t)(row - BHALF) * DDIM);
  const v4f* p4 = (const v4f*)src;
  v4f a = p4[lane * 2 + 0];
  v4f b = p4[lane * 2 + 1];

  float ss = 0.f;
#pragma unroll
  for (int i = 0; i < 4; ++i) ss += a[i] * a[i] + b[i] * b[i];
#pragma unroll
  for (int off = 16; off >= 1; off >>= 1) ss += __shfl_xor(ss, off, 32);

  const float inv = 1.0f / fmaxf(sqrtf(ss), 1e-8f);   // EPS from reference

  v8h o;
#pragma unroll
  for (int i = 0; i < 4; ++i) {
    o[i]     = (_Float16)(a[i] * inv);
    o[i + 4] = (_Float16)(b[i] * inv);
  }
  ((v8h*)(zn + (size_t)row * DDIM))[lane] = o;
}

// ---------------------------------------------------------------------------
// Kernel 2: fused Zn @ Zn^T + row-wise sum(exp(sim/T - 2)), diagonal excluded,
// positive-pair logit captured.
//   Wave  = 16-row strip x 1024-column chunk; A-strip (16x256 f16) register-
//           resident as 8 prepacked WMMA operands.
//   Block = 8 waves sharing the SAME column chunk (B tiles hit L0 after the
//           first wave) with 8 distinct row strips.
//   Inner loop software-pipelined: B operands for tile t+1 load while tile t
//   runs its 8 chained v_wmma_f32_16x16x32_f16 (unroll-2 ping-pong, no movs).
// ---------------------------------------------------------------------------
__global__ __launch_bounds__(256) void simclr_gemm_lse(
    const _Float16* __restrict__ zn,
    float* __restrict__ s_part, float* __restrict__ pos_part) {
  const int lane    = threadIdx.x & 31;
  const int wave    = threadIdx.x >> 5;
  const int chunk   = blockIdx.x >> 6;                 // 0..7  (shared by block)
  const int rowTile = (blockIdx.x & 63) * 8 + wave;    // 0..511
  const int rBase   = rowTile * 16;
  const int lanelo  = lane & 15;
  const int hl      = lane >> 4;               // lane half selects K sub-block

  // Preload entire A strip (rows rBase..rBase+15, K=0..255) into registers.
  const v8h* prA = (const v8h*)(zn + (size_t)(rBase + lanelo) * DDIM);
  v16h A[8];
#pragma unroll
  for (int kt = 0; kt < 8; ++kt)
    A[kt] = cat16(prA[kt * 4 + hl], prA[kt * 4 + 2 + hl]);

  float s[8], p[8];
#pragma unroll
  for (int g = 0; g < 8; ++g) { s[g] = 0.f; p[g] = 0.f; }

  const int colStart = chunk * (NROWS / NCHUNK);   // 1024 columns per chunk

  // Issue the 16 B128 loads for tile t (rows colStart+t*16 .. +15, full K).
  auto loadB = [&](v16h (&B)[8], int t) {
    const v8h* pr =
        (const v8h*)(zn + (size_t)(colStart + t * 16 + lanelo) * DDIM);
#pragma unroll
    for (int kt = 0; kt < 8; ++kt)
      B[kt] = cat16(pr[kt * 4 + hl], pr[kt * 4 + 2 + hl]);
  };

  // 8 chained WMMAs + fused logsumexp epilogue for one 16x16 tile.
  auto computeTile = [&](const v16h (&B)[8], int t) {
    v8f c = {};
#pragma unroll
    for (int kt = 0; kt < 8; ++kt)
      c = __builtin_amdgcn_wmma_f32_16x16x32_f16(
              /*neg_a=*/false, A[kt], /*neg_b=*/false, B[kt],
              /*c_mod=*/(short)0, c, /*reuse_a=*/false, /*reuse_b=*/false);

    const int col = colStart + t * 16 + lanelo;
#pragma unroll
    for (int g = 0; g < 8; ++g) {
      const int row = rBase + g + 8 * hl;
      float v = c[g] * 2.0f;                       // / TEMPERATURE (0.5)
      p[g] += (col == (row ^ BHALF)) ? v : 0.f;    // positive-pair logit
      // fixed max = 2.0 (sim/T <= 2): one exp per element, no online max
      s[g] += (col == row) ? 0.f : __expf(v - 2.0f);
    }
  };

  // Software pipeline, one tile deep, unroll-2 ping-pong.
  // (Final prefetches read <=32 rows past the chunk; for chunk 7 that lands
  //  inside the s_part workspace region -- allocated, values never used.)
  v16h B0[8], B1[8];
  loadB(B0, 0);
  for (int t = 0; t < 64; t += 2) {
    loadB(B1, t + 1);
    computeTile(B0, t);
    loadB(B0, t + 2);
    computeTile(B1, t + 1);
  }

  // Sum across the 16 lanes of each half-wave (columns live across lanes).
#pragma unroll
  for (int g = 0; g < 8; ++g) {
#pragma unroll
    for (int m = 8; m >= 1; m >>= 1) {
      s[g] += __shfl_xor(s[g], m, 32);
      p[g] += __shfl_xor(p[g], m, 32);
    }
  }
  if (lanelo == 0) {
#pragma unroll
    for (int g = 0; g < 8; ++g) {
      const int row = rBase + g + 8 * hl;
      s_part[(size_t)chunk * NROWS + row]   = s[g];
      pos_part[(size_t)chunk * NROWS + row] = p[g];
    }
  }
}

// ---------------------------------------------------------------------------
// Kernel 3: deterministic final reduction. Partial exp-sums combine by plain
// addition (shared fixed max). contrib = (2 + log(s)) - pos.
// ---------------------------------------------------------------------------
__global__ __launch_bounds__(1024) void simclr_reduce(
    const float* __restrict__ s_part, const float* __restrict__ pos_part,
    float* __restrict__ out) {
  __shared__ float red[1024];
  const int tid = threadIdx.x;
  float acc = 0.f;
  for (int r = tid; r < NROWS; r += 1024) {
    float s = 0.f, p = 0.f;
#pragma unroll
    for (int ch = 0; ch < NCHUNK; ++ch) {
      s += s_part[(size_t)ch * NROWS + r];
      p += pos_part[(size_t)ch * NROWS + r];
    }
    acc += (2.0f + __logf(s)) - p;
  }
  red[tid] = acc;
  __syncthreads();
  for (int off = 512; off >= 1; off >>= 1) {
    if (tid < off) red[tid] += red[tid + off];
    __syncthreads();
  }
  if (tid == 0) out[0] = red[0] / (float)NROWS;
}

// ---------------------------------------------------------------------------
extern "C" void kernel_launch(void* const* d_in, const int* in_sizes, int n_in,
                              void* d_out, int out_size, void* d_ws,
                              size_t ws_size, hipStream_t stream) {
  (void)in_sizes; (void)n_in; (void)out_size; (void)ws_size;

  const float* z_i = (const float*)d_in[0];
  const float* z_j = (const float*)d_in[1];

  char* ws = (char*)d_ws;
  _Float16* zn    = (_Float16*)ws;                                  // 4 MiB
  float* s_part   = (float*)(ws + (size_t)NROWS * DDIM * sizeof(_Float16));
  float* pos_part = s_part + (size_t)NCHUNK * NROWS;                // 256 KiB each

  simclr_normalize<<<NROWS / 8, 256, 0, stream>>>(z_i, z_j, zn);
  simclr_gemm_lse<<<(512 * NCHUNK) / 8, 256, 0, stream>>>(zn, s_part, pos_part);
  simclr_reduce<<<1, 1024, 0, stream>>>(s_part, pos_part, (float*)d_out);
}
